// SelfAttention_64862596104381
// MI455X (gfx1250) — 16-bit and 8-bit matrix operand order repaired and checked by reading, compiled, not run on hardware
//
#include <hip/hip_runtime.h>
#include <hip/hip_bf16.h>

typedef __bf16 bf16_t;
typedef bf16_t v16bf __attribute__((ext_vector_type(16)));
typedef bf16_t v8bf  __attribute__((ext_vector_type(8)));
typedef float  v8f   __attribute__((ext_vector_type(8)));
typedef int    v8i   __attribute__((ext_vector_type(8)));

#define BATCH 4
#define CCH   128
#define NPOS  4096
#define CQK   16
#define KPAD  32
#define LOG2E 1.4426950408889634f

__device__ __forceinline__ float xor16(float v) {
    int i = __builtin_bit_cast(int, v);
    int r = __builtin_amdgcn_permlanex16(i, i, 0x76543210, 0xfedcba98, false, false);
    return __builtin_bit_cast(float, r);
}

__global__ __launch_bounds__(256)
void proj_kernel(const float* __restrict__ x,
                 const float* __restrict__ Wq, const float* __restrict__ bq,
                 const float* __restrict__ Wk, const float* __restrict__ bk,
                 const float* __restrict__ Wv, const float* __restrict__ bv,
                 bf16_t* __restrict__ qT, bf16_t* __restrict__ kT,
                 unsigned char* __restrict__ vS)
{
    const int i = blockIdx.x * 256 + threadIdx.x;
    const int b = blockIdx.y;
    const int slice = blockIdx.z;
    const float* xb = x + (size_t)b * CCH * NPOS;

    float acc[32];
    if (slice == 0) {
        #pragma unroll
        for (int o = 0; o < 16; ++o) acc[o] = bq[o];
        #pragma unroll
        for (int o = 0; o < 16; ++o) acc[16 + o] = bk[o];
        #pragma unroll 4
        for (int c = 0; c < CCH; ++c) {
            const float xv = xb[(size_t)c * NPOS + i];
            #pragma unroll
            for (int o = 0; o < 16; ++o) acc[o]      += Wq[o * CCH + c] * xv;
            #pragma unroll
            for (int o = 0; o < 16; ++o) acc[16 + o] += Wk[o * CCH + c] * xv;
        }
        bf16_t* qrow = qT + ((size_t)b * NPOS + i) * CQK;
        bf16_t* krow = kT + ((size_t)b * NPOS + i) * KPAD;
        #pragma unroll
        for (int o = 0; o < 16; ++o) qrow[o] = (bf16_t)acc[o];
        #pragma unroll
        for (int o = 0; o < 16; ++o) krow[o] = (bf16_t)acc[16 + o];
        #pragma unroll
        for (int o = 16; o < 32; ++o) krow[o] = (bf16_t)0.0f;
    } else {
        const int c0 = (slice - 1) * 32;
        #pragma unroll
        for (int o = 0; o < 32; ++o) acc[o] = bv[c0 + o];
        #pragma unroll 4
        for (int c = 0; c < CCH; ++c) {
            const float xv = xb[(size_t)c * NPOS + i];
            #pragma unroll
            for (int o = 0; o < 32; ++o) acc[o] += Wv[(size_t)(c0 + o) * CCH + c] * xv;
        }
        const int blk = i >> 6;
        const int jp  = i & 63;
        const int g   = jp >> 5;
        const int sub = jp & 15;
        const int hi8 = (jp >> 4) & 1;
        const int q8  = sub;
        const int dwd = 4 * g + (q8 >> 2);
        const int byt = q8 & 3;
        #pragma unroll
        for (int o = 0; o < 32; ++o) {
            const int c = c0 + o;
            const int t = c >> 4;
            const int lane_row = (c & 15) + 16 * hi8;
            const int w = __builtin_amdgcn_cvt_pk_fp8_f32(acc[o], acc[o], 0, false);
            vS[((((size_t)(b * 8 + t) * 64 + blk) * 32) + lane_row) * 32 + dwd * 4 + byt]
                = (unsigned char)(w & 0xFF);
        }
    }
}

__global__ __launch_bounds__(128)
void attn_kernel(const bf16_t* __restrict__ qT, const bf16_t* __restrict__ kT,
                 const unsigned char* __restrict__ vS, const float* __restrict__ x,
                 const float* __restrict__ gamma, float* __restrict__ out)
{
    const int lane = threadIdx.x & 31;
    const int wave = threadIdx.x >> 5;
    const int ib   = blockIdx.x * 64 + wave * 16;
    const int b    = blockIdx.y;
    const int l16  = lane & 15;
    const int hi   = lane >> 4;
    const bool lo_half = (lane < 16);

    v16bf qb;
    #pragma unroll
    for (int m = 0; m < 16; ++m) qb[m] = (bf16_t)0.0f;
    if (lo_half) {
        qb = *(const v16bf*)(qT + ((size_t)b * NPOS + ib + lane) * CQK);
    }

    v8f zero8;
    #pragma unroll
    for (int r = 0; r < 8; ++r) zero8[r] = 0.0f;

    v8f acc[8];
    #pragma unroll
    for (int t = 0; t < 8; ++t) acc[t] = zero8;

    float ms_run = -3.0e38f;
    float l_run  = 0.0f;

    const bf16_t* kbase = kT + (size_t)b * NPOS * KPAD;

    #pragma unroll 2
    for (int jb = 0; jb < NPOS; jb += 64) {
        v8f s[4];
        #pragma unroll
        for (int u = 0; u < 4; ++u) {
            const bf16_t* krow = kbase + (size_t)(jb + u * 16 + l16) * KPAD;
            const v8bf t0 = *(const v8bf*)(krow + hi * 16);
            const v8bf t1 = *(const v8bf*)(krow + hi * 16 + 8);
            v16bf ka;
            #pragma unroll
            for (int m = 0; m < 8; ++m) { ka[m] = t0[m]; ka[8 + m] = t1[m]; }
            s[u] = __builtin_amdgcn_wmma_f32_16x16x32_bf16(
                       false, ka, false, qb, (short)0, zero8, false, false);
        }

        float mu[4];
        #pragma unroll
        for (int u = 0; u < 4; ++u) {
            const float a = fmaxf(s[u][0], s[u][1]);
            const float c = fmaxf(s[u][2], s[u][3]);
            const float d = fmaxf(s[u][4], s[u][5]);
            const float e = fmaxf(s[u][6], s[u][7]);
            mu[u] = fmaxf(fmaxf(a, c), fmaxf(d, e));
        }
        float mloc = fmaxf(fmaxf(mu[0], mu[1]), fmaxf(mu[2], mu[3]));
        mloc = fmaxf(mloc, xor16(mloc));
        const float ms_new = fmaxf(ms_run, mloc * LOG2E);
        const float corr   = __builtin_amdgcn_exp2f(ms_run - ms_new);

        float p[4][8];
        float lu[4];
        #pragma unroll
        for (int u = 0; u < 4; ++u) {
            #pragma unroll
            for (int r = 0; r < 8; ++r)
                p[u][r] = __builtin_amdgcn_exp2f(__builtin_fmaf(s[u][r], LOG2E, -ms_new));
            lu[u] = ((p[u][0] + p[u][1]) + (p[u][2] + p[u][3]))
                  + ((p[u][4] + p[u][5]) + (p[u][6] + p[u][7]));
        }
        const float lloc = (lu[0] + lu[1]) + (lu[2] + lu[3]);
        l_run = l_run * corr + lloc + xor16(lloc);
        ms_run = ms_new;

        #pragma unroll
        for (int t = 0; t < 8; ++t) acc[t] *= corr;

        v8i pb8;
        #pragma unroll
        for (int qd = 0; qd < 2; ++qd) {
            float fa[8], fb[8];
            #pragma unroll
            for (int m = 0; m < 8; ++m) {
                const float tsend = lo_half ? p[2 * qd + 1][m] : p[2 * qd][m];
                const float xt = xor16(tsend);
                fa[m] = lo_half ? p[2 * qd][m] : xt;
                fb[m] = lo_half ? xt : p[2 * qd + 1][m];
            }
            int w0 = __builtin_amdgcn_cvt_pk_fp8_f32(fa[0], fa[1], 0, false);
            w0     = __builtin_amdgcn_cvt_pk_fp8_f32(fa[2], fa[3], w0, true);
            int w1 = __builtin_amdgcn_cvt_pk_fp8_f32(fa[4], fa[5], 0, false);
            w1     = __builtin_amdgcn_cvt_pk_fp8_f32(fa[6], fa[7], w1, true);
            int w2 = __builtin_amdgcn_cvt_pk_fp8_f32(fb[0], fb[1], 0, false);
            w2     = __builtin_amdgcn_cvt_pk_fp8_f32(fb[2], fb[3], w2, true);
            int w3 = __builtin_amdgcn_cvt_pk_fp8_f32(fb[4], fb[5], 0, false);
            w3     = __builtin_amdgcn_cvt_pk_fp8_f32(fb[6], fb[7], w3, true);
            pb8[4 * qd + 0] = w0;
            pb8[4 * qd + 1] = w1;
            pb8[4 * qd + 2] = w2;
            pb8[4 * qd + 3] = w3;
        }

        const size_t blk = (size_t)(jb >> 6);
        #pragma unroll
        for (int t = 0; t < 8; ++t) {
            const v8i av = *(const v8i*)(vS +
                ((((size_t)(b * 8 + t) * 64 + blk) * 32) + lane) * 32);
            acc[t] = __builtin_amdgcn_wmma_f32_16x16x64_fp8_fp8(
                         av, pb8, (short)0, acc[t], false, false);
        }
    }

    const float inv_l = 1.0f / l_run;
    const float g = gamma[0];
    const float* xb = x   + (size_t)b * CCH * NPOS;
    float*       ob = out + (size_t)b * CCH * NPOS;
    const int i = ib + l16;
    #pragma unroll
    for (int t = 0; t < 8; ++t) {
        #pragma unroll
        for (int r = 0; r < 8; ++r) {
            const int c = t * 16 + r + 8 * hi;
            const size_t idx = (size_t)c * NPOS + i;
            ob[idx] = g * (acc[t][r] * inv_l) + xb[idx];
        }
    }
}

extern "C" void kernel_launch(void* const* d_in, const int* in_sizes, int n_in,
                              void* d_out, int out_size, void* d_ws, size_t ws_size,
                              hipStream_t stream) {
    (void)in_sizes; (void)n_in; (void)out_size; (void)ws_size;
    const float* x     = (const float*)d_in[0];
    const float* Wq    = (const float*)d_in[1];
    const float* bq    = (const float*)d_in[2];
    const float* Wk    = (const float*)d_in[3];
    const float* bk    = (const float*)d_in[4];
    const float* Wv    = (const float*)d_in[5];
    const float* bv    = (const float*)d_in[6];
    const float* gamma = (const float*)d_in[7];
    float* out = (float*)d_out;

    bf16_t*        qT = (bf16_t*)d_ws;
    bf16_t*        kT = (bf16_t*)((char*)d_ws + (size_t)(512 << 10));
    unsigned char* vS = (unsigned char*)d_ws + (size_t)(1536 << 10);

    dim3 g1(NPOS / 256, BATCH, 5);
    proj_kernel<<<g1, 256, 0, stream>>>(x, Wq, bq, Wk, bk, Wv, bv, qT, kT, vS);

    dim3 g2(NPOS / 64, BATCH);
    attn_kernel<<<g2, 128, 0, stream>>>(qT, kT, vS, x, gamma, out);
}
